// GraphEncoder_85718957293644
// MI455X (gfx1250) — matrix operand order repaired and checked by reading, not compiled, not run on hardware
//
#include <hip/hip_runtime.h>
#include <hip/hip_bf16.h>

// ===========================================================================
// GraphEncoder for MI455X (gfx1250, wave32, WMMA).
//
// Input index map (assumes recursive insertion-order flattening of the
// setup_inputs() dict):
//   0 node_features [B,N,32]        1 global_features [B,1,32]
//   2 edge_features (unused)
//   3 enc_w1  4 enc_b1  5 enc_ln_g  6 enc_ln_b  7 enc_w2  8 enc_b2
//   9 encode_token
//   10..27  attn_enc: wq,bq,wk,bk,wv,bv,wo,bo,q_ln,k_ln,ln1_g,ln1_b,w1,b1,w2,b2,ln2_g,ln2_b
//   28..45  attn_dec: (same order)
//   46 out_ln_g  47 out_ln_b
//   48+6i..53+6i (i=0..2): ln{i}_g, ln{i}_b, skip_w{i}, skip_b{i}, gcn_w{i}, gcn_b{i}
//   66 edge_list [B,E,2] i32   67 edge_mask [B,E] u8
//   68 node_mask [B,N] u8      69 current_agent_node_ind [B] i32
// ===========================================================================

#define Bn 4
#define Nn 8192        // 2^13
#define En 131072      // 2^17
#define Dm 128
#define EPSC 1e-6f

typedef __attribute__((ext_vector_type(16))) _Float16 v16h;
typedef __attribute__((ext_vector_type(8)))  float    v8f;

__device__ __forceinline__ float mishf(float x) {
  float sp = (x > 20.f) ? x : log1pf(expf(x));
  return x * tanhf(sp);
}

// ---- WMMA fragment builders (gfx1250 wave32 layouts, CDNA5 ISA) ----------
__device__ __forceinline__ v16h load_afrag(const _Float16* As, int lda,
                                           int r0, int k0, int lane) {
  int m = lane & 15, hh = lane >> 4;
  const _Float16* row = As + (r0 + m) * lda;
  v16h a;
#pragma unroll
  for (int v = 0; v < 8; ++v) {
    int kb = k0 + ((v & 4) ? 16 : 0) + (v & 3) * 2 + hh * 8;
    a[2 * v]     = row[kb];
    a[2 * v + 1] = row[kb + 1];
  }
  return a;
}

__device__ __forceinline__ v16h load_bfrag(const _Float16* Bs, int ldb,
                                           int k0, int n0, int lane) {
  int n = lane & 15, hh = lane >> 4;
  int kb = k0 + hh * 8;
  v16h b;
#pragma unroll
  for (int v = 0; v < 8; ++v) {
    int kk = kb + ((v & 4) ? 16 : 0) + (v & 3) * 2;
    b[2 * v]     = Bs[(kk)     * ldb + n0 + n];
    b[2 * v + 1] = Bs[(kk + 1) * ldb + n0 + n];
  }
  return b;
}

__device__ __forceinline__ v8f wmma_f16(v16h a, v16h b, v8f c) {
  return __builtin_amdgcn_wmma_f32_16x16x32_f16(false, a, false, b,
                                                (short)0, c, false, false);
}

// ===========================================================================
// Encoder stage 1: concat(nf, gf) [BN,64] @ enc_w1 [64,128] + b1, LN, mish.
// 64 rows per block, 8 waves, 4 output tiles of 16x16 per wave.
// ===========================================================================
__global__ __launch_bounds__(256)
void k_enc1(const float* __restrict__ nf, const float* __restrict__ gfv,
            const float* __restrict__ w1, const float* __restrict__ b1,
            const float* __restrict__ lng, const float* __restrict__ lnb,
            float* __restrict__ out) {
  __shared__ _Float16 wsld[64][132];   // 16.9 KB
  __shared__ _Float16 as[64][72];      //  9.2 KB
  __shared__ float    cs[64][132];     // 33.8 KB
  int tid = threadIdx.x;
  for (int i = tid; i < 64 * 128; i += 256)
    wsld[i >> 7][i & 127] = (_Float16)w1[i];
  int row0 = blockIdx.x * 64;
  for (int i = tid; i < 64 * 64; i += 256) {
    int r = i >> 6, c = i & 63;
    int g = row0 + r;
    int b = g >> 13;
    float val = (c < 32) ? nf[(size_t)g * 32 + c] : gfv[b * 32 + (c - 32)];
    as[r][c] = (_Float16)val;
  }
  __syncthreads();
  int wv = tid >> 5, lane = tid & 31, n = lane & 15, hg = lane >> 4;
  for (int t = wv; t < 32; t += 8) {
    int rt = t >> 3, ct = t & 7;
    v8f acc = {};
    for (int k0 = 0; k0 < 64; k0 += 32) {
      v16h a = load_afrag(&as[0][0], 72, rt * 16, k0, lane);
      v16h b = load_bfrag(&wsld[0][0], 132, k0, ct * 16, lane);
      acc = wmma_f16(a, b, acc);
    }
    float bb = b1[ct * 16 + n];
#pragma unroll
    for (int v = 0; v < 8; ++v)
      cs[rt * 16 + v + 8 * hg][ct * 16 + n] = acc[v] + bb;
  }
  __syncthreads();
  // LayerNorm(128) + mish per row; 8 rows per wave.
  for (int rr = 0; rr < 8; ++rr) {
    int r = wv * 8 + rr;
    float x0 = cs[r][lane * 4 + 0], x1 = cs[r][lane * 4 + 1];
    float x2 = cs[r][lane * 4 + 2], x3 = cs[r][lane * 4 + 3];
    float s  = x0 + x1 + x2 + x3;
    float sq = x0 * x0 + x1 * x1 + x2 * x2 + x3 * x3;
    for (int o = 16; o; o >>= 1) { s += __shfl_xor(s, o); sq += __shfl_xor(sq, o); }
    float m  = s * (1.f / 128.f);
    float va = fmaxf(sq * (1.f / 128.f) - m * m, 0.f);
    float rs = rsqrtf(va + EPSC);
    float* orow = out + (size_t)(row0 + r) * 128;
    int c = lane * 4;
    orow[c + 0] = mishf((x0 - m) * rs * lng[c + 0] + lnb[c + 0]);
    orow[c + 1] = mishf((x1 - m) * rs * lng[c + 1] + lnb[c + 1]);
    orow[c + 2] = mishf((x2 - m) * rs * lng[c + 2] + lnb[c + 2]);
    orow[c + 3] = mishf((x3 - m) * rs * lng[c + 3] + lnb[c + 3]);
  }
}

// ===========================================================================
// Generic [BN,128] x [128,128] GEMM, 64 rows per block, with optional input
// LayerNorm and optional per-head (16-wide) ln_scale epilogue (mode==1).
// ===========================================================================
__global__ __launch_bounds__(256)
void k_gemm128(const float* __restrict__ in, const float* __restrict__ lng,
               const float* __restrict__ lnb, const float* __restrict__ w,
               const float* __restrict__ bias, const float* __restrict__ g16,
               int mode, float* __restrict__ out) {
  __shared__ _Float16 wsld[128][132];  // 33.8 KB
  __shared__ _Float16 as[64][136];     // 17.4 KB
  int tid = threadIdx.x;
  for (int i = tid; i < 128 * 128; i += 256)
    wsld[i >> 7][i & 127] = (_Float16)w[i];
  int row0 = blockIdx.x * 64;
  int wv = tid >> 5, lane = tid & 31;
  for (int rr = 0; rr < 8; ++rr) {
    int r = wv * 8 + rr;
    const float* xr = in + (size_t)(row0 + r) * 128;
    int c = lane * 4;
    float x0 = xr[c + 0], x1 = xr[c + 1], x2 = xr[c + 2], x3 = xr[c + 3];
    if (lng) {
      float s  = x0 + x1 + x2 + x3;
      float sq = x0 * x0 + x1 * x1 + x2 * x2 + x3 * x3;
      for (int o = 16; o; o >>= 1) { s += __shfl_xor(s, o); sq += __shfl_xor(sq, o); }
      float m  = s * (1.f / 128.f);
      float va = fmaxf(sq * (1.f / 128.f) - m * m, 0.f);
      float rs = rsqrtf(va + EPSC);
      x0 = (x0 - m) * rs * lng[c + 0] + lnb[c + 0];
      x1 = (x1 - m) * rs * lng[c + 1] + lnb[c + 1];
      x2 = (x2 - m) * rs * lng[c + 2] + lnb[c + 2];
      x3 = (x3 - m) * rs * lng[c + 3] + lnb[c + 3];
    }
    as[r][c + 0] = (_Float16)x0; as[r][c + 1] = (_Float16)x1;
    as[r][c + 2] = (_Float16)x2; as[r][c + 3] = (_Float16)x3;
  }
  __syncthreads();
  int n = lane & 15, hg = lane >> 4;
  for (int t = wv; t < 32; t += 8) {
    int rt = t >> 3, ct = t & 7;
    v8f acc = {};
    for (int k0 = 0; k0 < 128; k0 += 32) {
      v16h a = load_afrag(&as[0][0], 136, rt * 16, k0, lane);
      v16h b = load_bfrag(&wsld[0][0], 132, k0, ct * 16, lane);
      acc = wmma_f16(a, b, acc);
    }
    float bb = bias[ct * 16 + n];
#pragma unroll
    for (int v = 0; v < 8; ++v) {
      float cval = acc[v] + bb;
      if (mode == 1) {  // per-head ln_scale: stats across 16-lane group
        float s = cval, sq = cval * cval;
#pragma unroll
        for (int o = 1; o < 16; o <<= 1) { s += __shfl_xor(s, o); sq += __shfl_xor(sq, o); }
        float m  = s * (1.f / 16.f);
        float va = fmaxf(sq * (1.f / 16.f) - m * m, 0.f);
        cval = (cval - m) * rsqrtf(va + EPSC) * g16[n];
      }
      out[(size_t)(row0 + rt * 16 + v + 8 * hg) * 128 + ct * 16 + n] = cval;
    }
  }
}

// ===========================================================================
// Query projection: q = ln_scale(split(row @ wq + bq)).  One block per batch.
// ===========================================================================
__global__ __launch_bounds__(128)
void k_qproj(const float* __restrict__ qsrc, long bstride, const int* __restrict__ qidx,
             const float* __restrict__ wq, const float* __restrict__ bq,
             const float* __restrict__ qln, float* __restrict__ qout) {
  int b = blockIdx.x, t = threadIdx.x;
  __shared__ float row[128], red[128];
  const float* src = qsrc + (size_t)b * bstride + (qidx ? (size_t)qidx[b] * 128 : 0);
  row[t] = src[t];
  __syncthreads();
  float acc = bq[t];
  for (int j = 0; j < 128; ++j) acc += row[j] * wq[j * 128 + t];
  red[t] = acc;
  __syncthreads();
  int h0 = (t >> 4) << 4;
  float s = 0.f, sq = 0.f;
  for (int j = 0; j < 16; ++j) { float x = red[h0 + j]; s += x; sq += x * x; }
  float m  = s * (1.f / 16.f);
  float va = fmaxf(sq * (1.f / 16.f) - m * m, 0.f);
  qout[b * 128 + t] = (acc - m) * rsqrtf(va + EPSC) * qln[t & 15];
}

// ===========================================================================
// Flash-style attention pooling: one block per (head, batch), online softmax
// over N keys with global_prefetch of the next k/v rows.
// ===========================================================================
__global__ __launch_bounds__(256)
void k_attn_pool(const float* __restrict__ q, const float* __restrict__ kb,
                 const float* __restrict__ vb, const unsigned char* __restrict__ nmask,
                 float* __restrict__ attn_out) {
  int h = blockIdx.x, b = blockIdx.y;
  int tid = threadIdx.x;
  __shared__ float qs[16];
  if (tid < 16) qs[tid] = q[b * 128 + h * 16 + tid];
  __syncthreads();
  float m = -3.0e38f, l = 0.f, acc[16];
#pragma unroll
  for (int d = 0; d < 16; ++d) acc[d] = 0.f;
  for (int n = tid; n < Nn; n += 256) {
    size_t base = ((size_t)(b * Nn + n)) * 128 + h * 16;
    if (n + 256 < Nn) {  // prefetch next tile (lowers to global_prefetch_b8)
      __builtin_prefetch(kb + base + 256 * 128, 0, 1);
      __builtin_prefetch(vb + base + 256 * 128, 0, 1);
    }
    float s = 0.f;
#pragma unroll
    for (int d = 0; d < 16; ++d) s += qs[d] * kb[base + d];
    s *= 0.25f;  // 1/sqrt(16)
    if (!nmask[b * Nn + n]) s = -1e9f;
    float mn = fmaxf(m, s);
    float e1 = expf(m - mn), e2 = expf(s - mn);
    l = l * e1 + e2;
#pragma unroll
    for (int d = 0; d < 16; ++d) acc[d] = acc[d] * e1 + e2 * vb[base + d];
    m = mn;
  }
  __shared__ float sm[256], sl[256], sa[256][17];
  sm[tid] = m; sl[tid] = l;
#pragma unroll
  for (int d = 0; d < 16; ++d) sa[tid][d] = acc[d];
  __syncthreads();
  for (int off = 128; off >= 1; off >>= 1) {
    if (tid < off) {
      float m2 = sm[tid + off], l2 = sl[tid + off];
      float mn = fmaxf(sm[tid], m2);
      float e1 = expf(sm[tid] - mn), e2 = expf(m2 - mn);
      sl[tid] = sl[tid] * e1 + l2 * e2;
#pragma unroll
      for (int d = 0; d < 16; ++d)
        sa[tid][d] = sa[tid][d] * e1 + sa[tid + off][d] * e2;
      sm[tid] = mn;
    }
    __syncthreads();
  }
  if (tid < 16) attn_out[b * 128 + h * 16 + tid] = sa[0][tid] / sl[0];
}

// ===========================================================================
// Encoder attention tail: ge = LN(tok + out@wo+bo); FFN; LN.  Block per batch.
// ===========================================================================
__global__ __launch_bounds__(128)
void k_attn_tail_enc(const float* __restrict__ tok, const float* __restrict__ aout,
                     const float* __restrict__ wo, const float* __restrict__ bo,
                     const float* __restrict__ ln1g, const float* __restrict__ ln1b,
                     const float* __restrict__ w1, const float* __restrict__ b1,
                     const float* __restrict__ w2, const float* __restrict__ b2,
                     const float* __restrict__ ln2g, const float* __restrict__ ln2b,
                     float* __restrict__ ge) {
  int b = blockIdx.x, t = threadIdx.x;
  __shared__ float o[128], xs[128], hs[128];
  o[t] = aout[b * 128 + t];
  __syncthreads();
  float y = bo[t];
  for (int j = 0; j < 128; ++j) y += o[j] * wo[j * 128 + t];
  y += tok[t];
  xs[t] = y;
  __syncthreads();
  float s = 0.f, sq = 0.f;
  for (int j = 0; j < 128; ++j) { s += xs[j]; sq += xs[j] * xs[j]; }
  float m  = s * (1.f / 128.f);
  float va = fmaxf(sq * (1.f / 128.f) - m * m, 0.f);
  float x = (y - m) * rsqrtf(va + EPSC) * ln1g[t] + ln1b[t];
  __syncthreads();
  xs[t] = x;
  __syncthreads();
  float h1 = b1[t];
  for (int j = 0; j < 128; ++j) h1 += xs[j] * w1[j * 128 + t];
  hs[t] = mishf(h1);
  __syncthreads();
  float y2 = b2[t];
  for (int j = 0; j < 128; ++j) y2 += hs[j] * w2[j * 128 + t];
  y2 += x;
  __syncthreads();
  xs[t] = y2;
  __syncthreads();
  s = 0.f; sq = 0.f;
  for (int j = 0; j < 128; ++j) { s += xs[j]; sq += xs[j] * xs[j]; }
  m  = s * (1.f / 128.f);
  va = fmaxf(sq * (1.f / 128.f) - m * m, 0.f);
  ge[b * 128 + t] = (y2 - m) * rsqrtf(va + EPSC) * ln2g[t] + ln2b[t];
}

// ===========================================================================
// Decoder attention tail: x = LN(dec + out@wo+bo, ln1); out = mish(LN(x)).
// ===========================================================================
__global__ __launch_bounds__(128)
void k_attn_tail_dec(const float* __restrict__ hbuf, const int* __restrict__ qidx,
                     const float* __restrict__ aout,
                     const float* __restrict__ wo, const float* __restrict__ bo,
                     const float* __restrict__ ln1g, const float* __restrict__ ln1b,
                     const float* __restrict__ olng, const float* __restrict__ olnb,
                     float* __restrict__ out) {
  int b = blockIdx.x, t = threadIdx.x;
  __shared__ float o[128], xs[128];
  o[t] = aout[b * 128 + t];
  __syncthreads();
  float y = bo[t];
  for (int j = 0; j < 128; ++j) y += o[j] * wo[j * 128 + t];
  y += hbuf[((size_t)b * Nn + qidx[b]) * 128 + t];
  xs[t] = y;
  __syncthreads();
  float s = 0.f, sq = 0.f;
  for (int j = 0; j < 128; ++j) { s += xs[j]; sq += xs[j] * xs[j]; }
  float m  = s * (1.f / 128.f);
  float va = fmaxf(sq * (1.f / 128.f) - m * m, 0.f);
  float x = (y - m) * rsqrtf(va + EPSC) * ln1g[t] + ln1b[t];
  __syncthreads();
  xs[t] = x;
  __syncthreads();
  s = 0.f; sq = 0.f;
  for (int j = 0; j < 128; ++j) { s += xs[j]; sq += xs[j] * xs[j]; }
  m  = s * (1.f / 128.f);
  va = fmaxf(sq * (1.f / 128.f) - m * m, 0.f);
  out[b * 128 + t] = mishf((x - m) * rsqrtf(va + EPSC) * olng[t] + olnb[t]);
}

// ===========================================================================
// Elementwise / graph kernels
// ===========================================================================
__global__ __launch_bounds__(256)
void k_add_ge(float* __restrict__ h, const float* __restrict__ ge) {
  size_t i = (size_t)blockIdx.x * 256 + threadIdx.x;   // < B*N*128
  int b = (int)(i >> 20);                              // N*128 = 2^20
  int c = (int)(i & 127);
  h[i] = mishf(h[i] + ge[b * 128 + c]);
}

__global__ __launch_bounds__(256)
void k_fill_deg(float* __restrict__ sdeg, float* __restrict__ rdeg) {
  int i = blockIdx.x * 256 + threadIdx.x;              // < B*N
  sdeg[i] = 1.f;  // self-edge contribution
  rdeg[i] = 1.f;
}

__global__ __launch_bounds__(256)
void k_deg(const int* __restrict__ elist, const unsigned char* __restrict__ emask,
           float* __restrict__ sdeg, float* __restrict__ rdeg) {
  int i = blockIdx.x * 256 + threadIdx.x;              // < B*E
  int b = i >> 17;                                     // E = 2^17
  if (!emask[i]) return;
  int s = elist[2 * i], r = elist[2 * i + 1];
  atomicAdd(&sdeg[b * Nn + s], 1.f);
  atomicAdd(&rdeg[b * Nn + r], 1.f);
}

__global__ __launch_bounds__(256)
void k_init_agg(const float* __restrict__ xg, const float* __restrict__ sdeg,
                float* __restrict__ agg) {
  size_t i = (size_t)blockIdx.x * 256 + threadIdx.x;   // < B*N*128
  int node = (int)(i >> 7);
  agg[i] = xg[i] * rsqrtf(fmaxf(sdeg[node], 1.f));     // self edge
}

// One wave32 per edge: gather 128 f32 from sender, atomic-add into receiver.
__global__ __launch_bounds__(256)
void k_scatter(const float* __restrict__ xg, const float* __restrict__ sdeg,
               const int* __restrict__ elist, const unsigned char* __restrict__ emask,
               float* __restrict__ agg) {
  int wv = threadIdx.x >> 5, lane = threadIdx.x & 31;
  int e = blockIdx.x * 8 + wv;                         // < B*E
  int b = e >> 17;
  if (!emask[e]) return;
  int s = elist[2 * e], r = elist[2 * e + 1];
  float inv = rsqrtf(fmaxf(sdeg[b * Nn + s], 1.f));
  const float* src = xg + (size_t)(b * Nn + s) * 128;
  float* dst = agg + (size_t)(b * Nn + r) * 128;
  int c = lane * 4;
  float4 v = *(const float4*)(src + c);
  atomicAdd(dst + c + 0, v.x * inv);
  atomicAdd(dst + c + 1, v.y * inv);
  atomicAdd(dst + c + 2, v.z * inv);
  atomicAdd(dst + c + 3, v.w * inv);
}

__global__ __launch_bounds__(256)
void k_gcn_final(const float* __restrict__ agg, const float* __restrict__ rdeg,
                 const float* __restrict__ skip, float* __restrict__ h) {
  size_t i = (size_t)blockIdx.x * 256 + threadIdx.x;   // < B*N*128
  int node = (int)(i >> 7);
  h[i] = mishf(agg[i] * rsqrtf(fmaxf(rdeg[node], 1.f)) + skip[i]);
}

// ===========================================================================
// Host-side orchestration
// ===========================================================================
extern "C" void kernel_launch(void* const* d_in, const int* in_sizes, int n_in,
                              void* d_out, int out_size, void* d_ws, size_t ws_size,
                              hipStream_t stream) {
  (void)in_sizes; (void)n_in; (void)out_size; (void)ws_size;
  auto F = [&](int i) { return (const float*)d_in[i]; };

  const float* nf      = F(0);
  const float* gfv     = F(1);
  const float* enc_w1  = F(3);  const float* enc_b1  = F(4);
  const float* enc_lng = F(5);  const float* enc_lnb = F(6);
  const float* enc_w2  = F(7);  const float* enc_b2  = F(8);
  const float* tok     = F(9);
  const float* out_lng = F(46); const float* out_lnb = F(47);
  const int*           elist = (const int*)d_in[66];
  const unsigned char* emask = (const unsigned char*)d_in[67];
  const unsigned char* nmask = (const unsigned char*)d_in[68];
  const int*           agent = (const int*)d_in[69];

  // attn param sub-indices (within a flattened attn_params dict)
  const int AE = 10, AD = 28;
  const int WQ=0,BQ=1,WK=2,BK=3,WV=4,BV=5,WO=6,BO=7,QLN=8,KLN=9,
            LN1G=10,LN1B=11,W1=12,B1=13,W2=14,B2=15,LN2G=16,LN2B=17;

  // Workspace layout (floats)
  float* ws    = (float*)d_ws;
  const size_t BN = (size_t)Bn * Nn;                  // 32768
  float* hbuf  = ws;                                  // [BN,128]
  float* bufA  = ws + BN * 128;                       // [BN,128]
  float* bufB  = ws + 2 * BN * 128;                   // [BN,128]
  float* bufC  = ws + 3 * BN * 128;                   // [BN,128]
  float* small = ws + 4 * BN * 128;
  float* qbuf  = small;                               // [B,128]
  float* aout  = small + Bn * 128;                    // [B,128]
  float* gebuf = small + 2 * Bn * 128;                // [B,128]
  float* sdeg  = small + 3 * Bn * 128;                // [B,N]
  float* rdeg  = sdeg + BN;                           // [B,N]

  const int gRows = (int)(BN / 64);                   // 512 GEMM blocks
  const int gElem = (int)(BN * 128 / 256);            // elementwise blocks
  const int gEdge = (Bn * En) / 256;
  const int gEdgeW = (Bn * En) / 8;

  // ---- node embed: Dense(64->128)+LN+mish, Dense(128->128) ----
  k_enc1<<<gRows, 256, 0, stream>>>(nf, gfv, enc_w1, enc_b1, enc_lng, enc_lnb, bufA);
  k_gemm128<<<gRows, 256, 0, stream>>>(bufA, nullptr, nullptr, enc_w2, enc_b2,
                                       nullptr, 0, hbuf);

  // ---- encoder attention pooling ----
  k_gemm128<<<gRows, 256, 0, stream>>>(hbuf, nullptr, nullptr, F(AE+WK), F(AE+BK),
                                       F(AE+KLN), 1, bufA);          // K (ln_scale)
  k_gemm128<<<gRows, 256, 0, stream>>>(hbuf, nullptr, nullptr, F(AE+WV), F(AE+BV),
                                       nullptr, 0, bufB);            // V
  k_qproj<<<Bn, 128, 0, stream>>>(tok, 0L, nullptr, F(AE+WQ), F(AE+BQ), F(AE+QLN), qbuf);
  k_attn_pool<<<dim3(8, Bn), 256, 0, stream>>>(qbuf, bufA, bufB, nmask, aout);
  k_attn_tail_enc<<<Bn, 128, 0, stream>>>(tok, aout, F(AE+WO), F(AE+BO),
                                          F(AE+LN1G), F(AE+LN1B),
                                          F(AE+W1), F(AE+B1), F(AE+W2), F(AE+B2),
                                          F(AE+LN2G), F(AE+LN2B), gebuf);
  k_add_ge<<<gElem, 256, 0, stream>>>(hbuf, gebuf);

  // ---- 3 GCN layers ----
  for (int i = 0; i < 3; ++i) {
    int base = 48 + 6 * i;
    const float* lng = F(base), *lnb = F(base + 1);
    const float* skw = F(base + 2), *skb = F(base + 3);
    const float* gw  = F(base + 4), *gb  = F(base + 5);
    k_gemm128<<<gRows, 256, 0, stream>>>(hbuf, lng, lnb, gw, gb, nullptr, 0, bufA);
    k_gemm128<<<gRows, 256, 0, stream>>>(hbuf, lng, lnb, skw, skb, nullptr, 0, bufB);
    k_fill_deg<<<(int)(BN / 256), 256, 0, stream>>>(sdeg, rdeg);
    k_deg<<<gEdge, 256, 0, stream>>>(elist, emask, sdeg, rdeg);
    k_init_agg<<<gElem, 256, 0, stream>>>(bufA, sdeg, bufC);
    k_scatter<<<gEdgeW, 256, 0, stream>>>(bufA, sdeg, elist, emask, bufC);
    k_gcn_final<<<gElem, 256, 0, stream>>>(bufC, rdeg, bufB, hbuf);
  }

  // ---- decoder attention ----
  k_gemm128<<<gRows, 256, 0, stream>>>(hbuf, nullptr, nullptr, F(AD+WK), F(AD+BK),
                                       F(AD+KLN), 1, bufA);
  k_gemm128<<<gRows, 256, 0, stream>>>(hbuf, nullptr, nullptr, F(AD+WV), F(AD+BV),
                                       nullptr, 0, bufB);
  k_qproj<<<Bn, 128, 0, stream>>>(hbuf, (long)Nn * 128, agent,
                                  F(AD+WQ), F(AD+BQ), F(AD+QLN), qbuf);
  k_attn_pool<<<dim3(8, Bn), 256, 0, stream>>>(qbuf, bufA, bufB, nmask, aout);
  k_attn_tail_dec<<<Bn, 128, 0, stream>>>(hbuf, agent, aout, F(AD+WO), F(AD+BO),
                                          F(AD+LN1G), F(AD+LN1B),
                                          out_lng, out_lnb, (float*)d_out);
}
